// TGCE_240518169112
// MI455X (gfx1250) — matrix operand order repaired and checked by reading, not compiled, not run on hardware
//
#include <hip/hip_runtime.h>
#include <hip/hip_bf16.h>

typedef __attribute__((ext_vector_type(16))) _Float16 v16h;
typedef __attribute__((ext_vector_type(8)))  float    v8f;

static constexpr int Bn = 8, Cn = 64, Hn = 64, Wn = 64;
static constexpr int HW = Hn * Wn;     // 4096
static constexpr int M  = Bn * HW;     // 32768 pixels total
static constexpr int C4 = 4 * Cn;      // 256
static constexpr int HID = 256;
static constexpr int Ltok = 16;

// ---------------------------------------------------------------- utilities
__global__ void k_cast_f16(const float* __restrict__ s, _Float16* __restrict__ d, int n) {
  int i = blockIdx.x * blockDim.x + threadIdx.x;
  if (i < n) d[i] = (_Float16)s[i];
}

// emb = mean_L(t);  x0 = relu(emb @ fc_w^T + fc_b)   -- one block per batch
__global__ void k_text_fc(const float* __restrict__ t, const float* __restrict__ fw,
                          const float* __restrict__ fb, float* __restrict__ x0) {
  __shared__ float emb[Cn];
  int b = blockIdx.x, c = threadIdx.x;
  float s = 0.f;
  for (int l = 0; l < Ltok; ++l) s += t[(b * Ltok + l) * Cn + c];
  emb[c] = s * (1.f / Ltok);
  __syncthreads();
  float acc = fb[c];
  for (int k = 0; k < Cn; ++k) acc += emb[k] * fw[c * Cn + k];
  x0[b * Cn + c] = fmaxf(acc, 0.f);
}

// broadcast (B,C) -> channels-last map (B,H,W,C)
__global__ void k_bcast(const float* __restrict__ x0, float* __restrict__ map) {
  int p = blockIdx.x, c = threadIdx.x;
  map[p * Cn + c] = x0[(p >> 12) * Cn + c];
}

// BN(eval) + cast to f16 (A matrix for GEMM1)
__global__ void k_bncast(const float* __restrict__ map, const float* __restrict__ g,
                         const float* __restrict__ bb, const float* __restrict__ mm,
                         const float* __restrict__ vv, _Float16* __restrict__ Xh) {
  int i = blockIdx.x * blockDim.x + threadIdx.x;  // M*Cn elems
  int c = i & (Cn - 1);
  float sc = g[c] * rsqrtf(vv[c] + 1e-5f);
  Xh[i] = (_Float16)((map[i] - mm[c]) * sc + bb[c]);
}

// ------------------------------------------------------------- WMMA GEMMs
// Y(MxN,f16) = relu(A(MxK,f16) @ Bw(NxK,f16)^T + bias)
__global__ void __launch_bounds__(256)
k_gemm_relu(const _Float16* __restrict__ A, const _Float16* __restrict__ Bw,
            const float* __restrict__ bias, _Float16* __restrict__ Y, int K, int N) {
  int lane = threadIdx.x & 31, wave = threadIdx.x >> 5;
  int hi = lane >> 4, ln = lane & 15;
  int tm = blockIdx.x * 32 + (wave >> 2) * 16;
  int tn = blockIdx.y * 64 + (wave & 3) * 16;
  int m = tm + ln, n = tn + ln;
  v8f acc = {};
  for (int k = 0; k < K; k += 32) {
    v16h a, b;
    const _Float16* ap = A + m * K + k + hi * 8;
    const _Float16* bp = Bw + n * K + k + hi * 8;
#pragma unroll
    for (int j = 0; j < 8; ++j) { a[j] = ap[j]; a[8 + j] = ap[16 + j]; }
#pragma unroll
    for (int j = 0; j < 8; ++j) { b[j] = bp[j]; b[8 + j] = bp[16 + j]; }
    acc = __builtin_amdgcn_wmma_f32_16x16x32_f16(false, a, false, b, (short)0, acc, false, false);
  }
#pragma unroll
  for (int r = 0; r < 8; ++r) {
    int row = tm + hi * 8 + r, col = tn + ln;
    Y[row * N + col] = (_Float16)fmaxf(acc[r] + bias[col], 0.f);
  }
}

// Ymap(MxN,f32) += A(MxK,f16) @ Bw(NxK,f16)^T + bias   (residual, in place)
__global__ void __launch_bounds__(256)
k_gemm_res(const _Float16* __restrict__ A, const _Float16* __restrict__ Bw,
           const float* __restrict__ bias, float* __restrict__ Ymap, int K, int N) {
  int lane = threadIdx.x & 31, wave = threadIdx.x >> 5;
  int hi = lane >> 4, ln = lane & 15;
  int tm = blockIdx.x * 32 + (wave >> 2) * 16;
  int tn = blockIdx.y * 64 + (wave & 3) * 16;
  int m = tm + ln, n = tn + ln;
  v8f acc = {};
  for (int k = 0; k < K; k += 32) {
    v16h a, b;
    const _Float16* ap = A + m * K + k + hi * 8;
    const _Float16* bp = Bw + n * K + k + hi * 8;
#pragma unroll
    for (int j = 0; j < 8; ++j) { a[j] = ap[j]; a[8 + j] = ap[16 + j]; }
#pragma unroll
    for (int j = 0; j < 8; ++j) { b[j] = bp[j]; b[8 + j] = bp[16 + j]; }
    acc = __builtin_amdgcn_wmma_f32_16x16x32_f16(false, a, false, b, (short)0, acc, false, false);
  }
#pragma unroll
  for (int r = 0; r < 8; ++r) {
    int row = tm + hi * 8 + r, col = tn + ln;
    Ymap[row * N + col] = acc[r] + bias[col] + Ymap[row * N + col];
  }
}

// roll(W,+1) then depthwise 3x3 (zero pad) + bias + relu, channels-last
__global__ void k_dwconv(const _Float16* __restrict__ Y1, const float* __restrict__ dk,
                         const float* __restrict__ db, _Float16* __restrict__ Y1d) {
  int p = blockIdx.x, o = threadIdx.x;
  int b = p >> 12, h = (p >> 6) & 63, w = p & 63;
  float acc = db[o];
#pragma unroll
  for (int dh = -1; dh <= 1; ++dh) {
    int hh = h + dh; if ((unsigned)hh >= 64u) continue;
#pragma unroll
    for (int dx = -1; dx <= 1; ++dx) {
      int wx = w + dx; if ((unsigned)wx >= 64u) continue;
      int wsrc = (wx - 1) & 63;  // rolled source column
      acc += (float)Y1[(((b * 64 + hh) * 64 + wsrc) * 256) + o] * dk[o * 9 + (dh + 1) * 3 + (dx + 1)];
    }
  }
  Y1d[p * 256 + o] = (_Float16)fmaxf(acc, 0.f);
}

// T = A*B*AB, normalized; emit f32 (gather source) and f16 (WMMA B)
__global__ void k_make_tn(const float* __restrict__ mA, const float* __restrict__ mB,
                          const float* __restrict__ mAB, float* __restrict__ tn32,
                          _Float16* __restrict__ tn16) {
  __shared__ float sh[Cn];
  int p = blockIdx.x, c = threadIdx.x, i = p * Cn + c;
  float t = mA[i] * mB[i] * mAB[i];
  sh[c] = t * t; __syncthreads();
  for (int off = 32; off >= 1; off >>= 1) { if (c < off) sh[c] += sh[c + off]; __syncthreads(); }
  float val = t / (sqrtf(sh[0]) + 1e-6f);
  tn32[i] = val; tn16[i] = (_Float16)val;
}

// transpose V NCHW -> (B,HW,C) f32, plus normalized f16 copy
__global__ void k_make_vn(const float* __restrict__ V, float* __restrict__ Vc,
                          _Float16* __restrict__ vn16) {
  __shared__ float sh[Cn];
  int p = blockIdx.x, c = threadIdx.x;
  int b = p >> 12, hw = p & 4095;
  float v = V[(b * Cn + c) * HW + hw];
  Vc[p * Cn + c] = v;
  sh[c] = v * v; __syncthreads();
  for (int off = 32; off >= 1; off >>= 1) { if (c < off) sh[c] += sh[c + off]; __syncthreads(); }
  vn16[p * Cn + c] = (_Float16)(v / (sqrtf(sh[0]) + 1e-6f));
}

// fused sim GEMM + row argmax: wave owns 16 rows, streams 256 col-tiles
__global__ void __launch_bounds__(256)
k_sim_argmax(const _Float16* __restrict__ vn, const _Float16* __restrict__ tn,
             int* __restrict__ idx) {
  int lane = threadIdx.x & 31, wave = threadIdx.x >> 5;
  int rg = blockIdx.x * 8 + wave;      // 16-row group
  int b = rg >> 8;
  int m0 = (rg & 255) * 16;
  int hi = lane >> 4, ln = lane & 15;
  const _Float16* Ab = vn + (size_t)b * HW * Cn;
  const _Float16* Tb = tn + (size_t)b * HW * Cn;
  v16h a0, a1;
  {
    const _Float16* ap = Ab + (m0 + ln) * Cn + hi * 8;
#pragma unroll
    for (int j = 0; j < 8; ++j) {
      a0[j] = ap[j];      a0[8 + j] = ap[16 + j];
      a1[j] = ap[32 + j]; a1[8 + j] = ap[48 + j];
    }
  }
  float best[8]; int bidx[8];
#pragma unroll
  for (int r = 0; r < 8; ++r) { best[r] = -3.4e38f; bidx[r] = 0; }
  for (int ct = 0; ct < HW / 16; ++ct) {
    int n = ct * 16 + ln;
    const _Float16* bp = Tb + n * Cn + hi * 8;
    v16h b0, b1;
#pragma unroll
    for (int j = 0; j < 8; ++j) { b0[j] = bp[j]; b0[8 + j] = bp[16 + j]; b1[j] = bp[32 + j]; b1[8 + j] = bp[48 + j]; }
    v8f acc = {};
    acc = __builtin_amdgcn_wmma_f32_16x16x32_f16(false, a0, false, b0, (short)0, acc, false, false);
    acc = __builtin_amdgcn_wmma_f32_16x16x32_f16(false, a1, false, b1, (short)0, acc, false, false);
#pragma unroll
    for (int r = 0; r < 8; ++r) {
      float v = acc[r];
      if (v > best[r]) { best[r] = v; bidx[r] = ct * 16 + ln; }
    }
  }
#pragma unroll
  for (int r = 0; r < 8; ++r) {
    for (int off = 8; off >= 1; off >>= 1) {  // reduce across the 16 lanes of a row
      float ov = __shfl_xor(best[r], off, 16);
      int   oi = __shfl_xor(bidx[r], off, 16);
      if (ov > best[r] || (ov == best[r] && oi < bidx[r])) { best[r] = ov; bidx[r] = oi; }
    }
    if (ln == 0) idx[b * HW + m0 + hi * 8 + r] = bidx[r];
  }
}

__global__ void k_gather(const float* __restrict__ tn32, const int* __restrict__ idx,
                         float* __restrict__ Tre) {
  int p = blockIdx.x, c = threadIdx.x;
  int b = p >> 12, j = idx[p];
  Tre[p * Cn + c] = tn32[((size_t)b * HW + j) * Cn + c];
}

// recursive domain transform along W (one block per (b,h); lane = channel)
__global__ void k_dt_row(const float* __restrict__ Vc, const float* __restrict__ Tre,
                         float* __restrict__ Vo) {
  __shared__ float sd[Cn], sv[Cn], st[Cn];
  int bh = blockIdx.x, c = threadIdx.x;
  const float* vr = Vc + (size_t)bh * Wn * Cn;
  const float* tr = Tre + (size_t)bh * Wn * Cn;
  float* orow = Vo + (size_t)bh * Wn * Cn;
  float prev = vr[c];
  orow[c] = prev;
  for (int w = 1; w < Wn; ++w) {
    float vc = vr[w * Cn + c], tc = tr[(w - 1) * Cn + c];
    sd[c] = vc * tc; sv[c] = vc * vc; st[c] = tc * tc;
    __syncthreads();
    for (int off = 32; off >= 1; off >>= 1) {
      if (c < off) { sd[c] += sd[c + off]; sv[c] += sv[c + off]; st[c] += st[c + off]; }
      __syncthreads();
    }
    float cosd = 1.f - sd[0] / (sqrtf(sv[0]) * sqrtf(st[0]) + 1e-8f);
    float a = __expf(-cosd);
    prev = (1.f - a) * vc + a * prev;
    orow[w * Cn + c] = prev;
    __syncthreads();
  }
}

// recursive domain transform along H (one block per (b,w))
__global__ void k_dt_col(const float* __restrict__ Vi, const float* __restrict__ Tre,
                         float* __restrict__ Vo) {
  __shared__ float sd[Cn], sv[Cn], st[Cn];
  int bw = blockIdx.x, c = threadIdx.x;
  int b = bw >> 6, w = bw & 63;
  size_t base = (size_t)b * Hn * Wn * Cn + (size_t)w * Cn;
  float prev = Vi[base + c];
  Vo[base + c] = prev;
  for (int h = 1; h < Hn; ++h) {
    size_t iv = base + (size_t)h * Wn * Cn;
    float vc = Vi[iv + c], tc = Tre[base + (size_t)(h - 1) * Wn * Cn + c];
    sd[c] = vc * tc; sv[c] = vc * vc; st[c] = tc * tc;
    __syncthreads();
    for (int off = 32; off >= 1; off >>= 1) {
      if (c < off) { sd[c] += sd[c + off]; sv[c] += sv[c + off]; st[c] += st[c + off]; }
      __syncthreads();
    }
    float cosd = 1.f - sd[0] / (sqrtf(sv[0]) * sqrtf(st[0]) + 1e-8f);
    float a = __expf(-cosd);
    prev = (1.f - a) * vc + a * prev;
    Vo[iv + c] = prev;
    __syncthreads();
  }
}

// cosine distances + two 1->256->1 MLPs + sigmoid (one thread per pixel)
__global__ void k_weight(const float* __restrict__ Vdt, const float* __restrict__ Tre,
                         const float* __restrict__ tvw1, const float* __restrict__ tvb1,
                         const float* __restrict__ tvw2, const float* __restrict__ tvb2,
                         const float* __restrict__ ttw1, const float* __restrict__ ttb1,
                         const float* __restrict__ ttw2, const float* __restrict__ ttb2,
                         float* __restrict__ wgt) {
  int p = blockIdx.x * blockDim.x + threadIdx.x;
  int pin = p & 4095;
  const float* v = Vdt + (size_t)p * Cn;
  const float* t = Tre + (size_t)p * Cn;
  float d0 = 0, nv = 0, nt = 0;
  for (int c = 0; c < Cn; ++c) { float a = v[c], b = t[c]; d0 += a * b; nv += a * a; nt += b * b; }
  float dtv = 1.f - d0 / (sqrtf(nv) * sqrtf(nt) + 1e-8f);
  float dtt = 0.f;
  if (pin != 4095) {
    const float* t2 = t + Cn;
    float d1 = 0, n1 = 0, n2 = 0;
    for (int c = 0; c < Cn; ++c) { float a = t[c], b = t2[c]; d1 += a * b; n1 += a * a; n2 += b * b; }
    dtt = 1.f - d1 / (sqrtf(n1) * sqrtf(n2) + 1e-8f);
  }
  float s = tvb2[0] + ttb2[0];
  for (int k = 0; k < HID; ++k) {
    s += fmaxf(dtv * tvw1[k] + tvb1[k], 0.f) * tvw2[k];
    s += fmaxf(dtt * ttw1[k] + ttb1[k], 0.f) * ttw2[k];
  }
  wgt[p] = 1.f / (1.f + __expf(-s));
}

// output: (B,C,H,W) = Vdt(B,H,W,C) * w(B,1,H,W)
__global__ void k_out(const float* __restrict__ Vdt, const float* __restrict__ wgt,
                      float* __restrict__ out) {
  int i = blockIdx.x * blockDim.x + threadIdx.x;
  int b = i >> 18, c = (i >> 12) & 63, h = (i >> 6) & 63, w = i & 63;
  out[i] = Vdt[((((size_t)b * 64 + h) * 64 + w) * 64) + c] * wgt[b * HW + h * 64 + w];
}

// ---------------------------------------------------------------- host
extern "C" void kernel_launch(void* const* d_in, const int* in_sizes, int n_in,
                              void* d_out, int out_size, void* d_ws, size_t ws_size,
                              hipStream_t stream) {
  (void)in_sizes; (void)n_in; (void)ws_size;
  const float* V   = (const float*)d_in[0];
  const float* tA  = (const float*)d_in[1];
  const float* tB  = (const float*)d_in[2];
  const float* tAB = (const float*)d_in[3];
  // params pytree (sorted keys): mlp_tt(b1,b2,w1,w2), mlp_tv, text_A, text_AB, text_B
  const float* tt_b1 = (const float*)d_in[4];
  const float* tt_b2 = (const float*)d_in[5];
  const float* tt_w1 = (const float*)d_in[6];
  const float* tt_w2 = (const float*)d_in[7];
  const float* tv_b1 = (const float*)d_in[8];
  const float* tv_b2 = (const float*)d_in[9];
  const float* tv_w1 = (const float*)d_in[10];
  const float* tv_w2 = (const float*)d_in[11];
  const int texBase[3] = {12, 96, 54};                 // text_A, text_B, text_AB
  const float* texTok[3] = {tA, tB, tAB};

  char* ws = (char*)d_ws;
  float*    map0 = (float*)(ws + 0);
  float*    map1 = (float*)(ws + 8388608);
  float*    map2 = (float*)(ws + 16777216);
  float*    maps[3] = {map0, map1, map2};
  // phase-1 (block processing) scratch, reused by phase 2 afterwards
  _Float16* Xh   = (_Float16*)(ws + 25165824);         // 4 MB
  _Float16* Y1   = (_Float16*)(ws + 29360128);         // 16 MB
  _Float16* Y1d  = (_Float16*)(ws + 46137344);         // 16 MB
  _Float16* w1h  = (_Float16*)(ws + 62914560);
  _Float16* w2h  = (_Float16*)(ws + 62947328);
  float*    x0   = (float*)  (ws + 62980096);          // 3*B*C
  // phase-2 (overlaps Xh/Y1/Y1d, which are dead by then)
  float*    tn32 = (float*)   (ws + 25165824);
  _Float16* tn16 = (_Float16*)(ws + 33554432);
  _Float16* vn16 = (_Float16*)(ws + 37748736);
  float*    Vc   = (float*)   (ws + 41943040);
  float*    Tre  = (float*)   (ws + 50331648);
  int*      idx  = (int*)     (ws + 58720256);
  float*    wgt  = (float*)   (ws + 58851328);

  // ---- text paths: FC -> broadcast -> 4 CycleMLP blocks each
  for (int p = 0; p < 3; ++p) {
    const float* fcb = (const float*)d_in[texBase[p] + 40];
    const float* fcw = (const float*)d_in[texBase[p] + 41];
    k_text_fc<<<Bn, Cn, 0, stream>>>(texTok[p], fcw, fcb, x0 + p * Bn * Cn);
    k_bcast<<<M, Cn, 0, stream>>>(x0 + p * Bn * Cn, maps[p]);
    for (int blk = 0; blk < 4; ++blk) {
      int base = texBase[p] + blk * 10;  // b1,b2,bn_b,bn_g,bn_m,bn_v,dw,dwb,w1,w2
      const float* b1  = (const float*)d_in[base + 0];
      const float* b2  = (const float*)d_in[base + 1];
      const float* bnb = (const float*)d_in[base + 2];
      const float* bng = (const float*)d_in[base + 3];
      const float* bnm = (const float*)d_in[base + 4];
      const float* bnv = (const float*)d_in[base + 5];
      const float* dwk = (const float*)d_in[base + 6];
      const float* dwb = (const float*)d_in[base + 7];
      const float* w1  = (const float*)d_in[base + 8];
      const float* w2  = (const float*)d_in[base + 9];
      k_cast_f16<<<64, 256, 0, stream>>>(w1, w1h, C4 * Cn);
      k_cast_f16<<<64, 256, 0, stream>>>(w2, w2h, Cn * C4);
      k_bncast<<<M * Cn / 256, 256, 0, stream>>>(maps[p], bng, bnb, bnm, bnv, Xh);
      dim3 g1(M / 32, C4 / 64);
      k_gemm_relu<<<g1, 256, 0, stream>>>(Xh, w1h, b1, Y1, Cn, C4);
      k_dwconv<<<M, C4, 0, stream>>>(Y1, dwk, dwb, Y1d);
      dim3 g2(M / 32, Cn / 64);
      k_gemm_res<<<g2, 256, 0, stream>>>(Y1d, w2h, b2, maps[p], C4, Cn);
    }
  }

  // ---- T, normalization, retrieval, domain transform, gating
  k_make_tn<<<M, Cn, 0, stream>>>(map0, map1, map2, tn32, tn16);
  k_make_vn<<<M, Cn, 0, stream>>>(V, Vc, vn16);
  k_sim_argmax<<<M / 16 / 8, 256, 0, stream>>>(vn16, tn16, idx);
  k_gather<<<M, Cn, 0, stream>>>(tn32, idx, Tre);
  float* Vrow = map0;  // maps are dead after k_make_tn
  float* Vdt  = map1;
  k_dt_row<<<Bn * Hn, Cn, 0, stream>>>(Vc, Tre, Vrow);
  k_dt_col<<<Bn * Wn, Cn, 0, stream>>>(Vrow, Tre, Vdt);
  k_weight<<<M / 256, 256, 0, stream>>>(Vdt, Tre, tv_w1, tv_b1, tv_w2, tv_b2,
                                        tt_w1, tt_b1, tt_w2, tt_b2, wgt);
  k_out<<<out_size / 256, 256, 0, stream>>>(Vdt, wgt, (float*)d_out);
}
